// GraphAttentionLayer_57552561766401
// MI455X (gfx1250) — matrix operand order repaired and checked by reading, not compiled, not run on hardware
//
#include <hip/hip_runtime.h>
#include <hip/hip_bf16.h>

// ---------------------------------------------------------------------------
// GraphAttentionLayer for MI455X (gfx1250, wave32, WMMA).
//
//  * Edge MLP (We, Wa) folded algebraically into two [256,8] node-level
//    matrices (We1@Wa, We2@Wa): all dense math becomes node-level WMMA GEMMs.
//  * One fused GEMM produces [q|k|v|a1|a2|pad] = x @ Wcat
//    ([40000,256] x [256,800]) via v_wmma_f32_16x16x32_f16 (f16 in, f32 acc).
//    Each wave computes a 16x32 output tile (2 WMMAs per k-step) so the
//    A fragment load+convert is amortized over 2 MMAs.
//  * Thin per-edge passes (score+max, exp+sum, weighted scatter) gather only
//    from node tables (~128 MB -> resident in the 192 MB L2).
//  * Final out = acc @ Wo + bo via the same WMMA GEMM.
// ---------------------------------------------------------------------------

typedef __attribute__((ext_vector_type(16))) _Float16 v16h;
typedef __attribute__((ext_vector_type(8)))  float    v8f;

#define B_      4
#define N_      10000
#define E_      160000
#define HEADS_  8
#define BN_     (B_ * N_)
#define BE_     (B_ * E_)
#define NCAT    800          // 256(q)+256(k)+256(v)+8(a1)+8(a2)+16(pad) = 50*16
#define NT_CAT  50

// ---------------------------------------------------------------------------
// Repack weights (f32 row-major [K=256, N]) into a 16-bit WMMA B-matrix
// 32x16 fragment layout in contiguous k order (k = 16*half + j), pre-converted to f16:
//   tile (nt, kt): lane = nn + (kk & 16), element j = kk & 15
//   flat: packed[((nt*8+kt)*32 + lane)*16 + j]
// Also composes We1@Wa / We2@Wa into columns 768..783 and builds fused biases.
// ---------------------------------------------------------------------------
__global__ void repack_weights(const float* __restrict__ Wq, const float* __restrict__ Wk,
                               const float* __restrict__ Wv, const float* __restrict__ We,
                               const float* __restrict__ Wa, const float* __restrict__ Wo,
                               const float* __restrict__ bq, const float* __restrict__ bk,
                               const float* __restrict__ bv, const float* __restrict__ be,
                               const float* __restrict__ ba,
                               _Float16* __restrict__ packW, _Float16* __restrict__ packWo,
                               float* __restrict__ bcat, float* __restrict__ cbias) {
  int idx = blockIdx.x * blockDim.x + threadIdx.x;
  const int TOT1 = NCAT * 256;
  if (idx < TOT1) {
    int n = idx >> 8;          // output column 0..799
    int k = idx & 255;         // reduction row 0..255
    float val;
    if (n < 256)      val = Wq[k * 256 + n];
    else if (n < 512) val = Wk[k * 256 + (n - 256)];
    else if (n < 768) val = Wv[k * 256 + (n - 512)];
    else if (n < 776) {        // (We[:256] @ Wa)[k, n-768]
      int j = n - 768; float s = 0.f;
      for (int c = 0; c < 64; ++c) s += We[k * 64 + c] * Wa[c * 8 + j];
      val = s;
    } else if (n < 784) {      // (We[256:] @ Wa)[k, n-776]
      int j = n - 776; float s = 0.f;
      for (int c = 0; c < 64; ++c) s += We[(k + 256) * 64 + c] * Wa[c * 8 + j];
      val = s;
    } else {
      val = 0.f;               // pad columns 784..799
    }
    int nt = n >> 4, nn = n & 15, kt = k >> 5, kk = k & 31;
    int lane = nn + (kk & 16);
    packW[(((size_t)(nt * 8 + kt) * 32 + lane) << 4) + (kk & 15)] = (_Float16)val;
    if (k == 0)
      bcat[n] = (n < 256) ? bq[n] : (n < 512) ? bk[n - 256] : (n < 768) ? bv[n - 512] : 0.f;
    if (k == 1 && n < 8) {     // cbias = be @ Wa + ba
      float s = ba[n];
      for (int c = 0; c < 64; ++c) s += be[c] * Wa[c * 8 + n];
      cbias[n] = s;
    }
  } else if (idx < TOT1 + 256 * 256) {
    int i2 = idx - TOT1;
    int n = i2 >> 8, k = i2 & 255;
    float val = Wo[k * 256 + n];
    int nt = n >> 4, nn = n & 15, kt = k >> 5, kk = k & 31;
    int lane = nn + (kk & 16);
    packWo[(((size_t)(nt * 8 + kt) * 32 + lane) << 4) + (kk & 15)] = (_Float16)val;
  }
}

// ---------------------------------------------------------------------------
// Y[M=grid.x*16, ldY] = A[M,256](f32->f16) @ packedB + bias, via WMMA.
// One wave per 16x32 output tile (N-tile pair); 8 waves/block share an M-tile.
// A fragment (16-bit A 16x32): lane l (m=l&15, h=l>>4) holds K = h*16+{0..7}
// and h*16+8+{0..7} -> one contiguous 16-float run, converted to f16, then
// reused by two back-to-back WMMAs (independent accumulators).
// ---------------------------------------------------------------------------
__global__ __launch_bounds__(256) void gemm_xw_wmma(const float* __restrict__ A,
                                                    const _Float16* __restrict__ packB,
                                                    const float* __restrict__ bias,
                                                    float* __restrict__ Y,
                                                    int ldY, int numPairs) {
  int wave = threadIdx.x >> 5;
  int lane = threadIdx.x & 31;
  int pair = blockIdx.y * 8 + wave;
  if (pair >= numPairs) return;              // wave-uniform: EXEC all-1s for WMMA
  int nt0 = pair * 2;
  int mbase = blockIdx.x * 16;
  int m = lane & 15;
  int half = lane >> 4;
  const float* arow = A + (size_t)(mbase + m) * 256 + half * 16;
  const _Float16* bb0 = packB + (size_t)nt0 * 8 * 512 + (size_t)lane * 16;
  const _Float16* bb1 = bb0 + (size_t)8 * 512;
  v8f acc0 = {}, acc1 = {};
#pragma unroll
  for (int kt = 0; kt < 8; ++kt) {
    const float* ap = arow + kt * 32;
    v16h a;
#pragma unroll
    for (int i = 0; i < 8; ++i) {
      a[i]     = (_Float16)ap[i];
      a[i + 8] = (_Float16)ap[8 + i];
    }
    v16h b0 = *(const v16h*)(bb0 + (size_t)kt * 512);
    v16h b1 = *(const v16h*)(bb1 + (size_t)kt * 512);
    acc0 = __builtin_amdgcn_wmma_f32_16x16x32_f16(false, a, false, b0, (short)0, acc0,
                                                  false, false);
    acc1 = __builtin_amdgcn_wmma_f32_16x16x32_f16(false, a, false, b1, (short)0, acc1,
                                                  false, false);
  }
  int col0 = nt0 * 16 + m;
  float bv0 = bias[col0];
  float bv1 = bias[col0 + 16];
  float* yc = Y + (size_t)(mbase + half * 8) * ldY + col0;
#pragma unroll
  for (int r = 0; r < 8; ++r) {
    yc[(size_t)r * ldY]      = acc0[r] + bv0;
    yc[(size_t)r * ldY + 16] = acc1[r] + bv1;
  }
}

// ---------------------------------------------------------------------------
__global__ void init_buffers(float* __restrict__ smax, float* __restrict__ denom,
                             float* __restrict__ outacc) {
  int i = blockIdx.x * blockDim.x + threadIdx.x;
  if (i < BN_ * 256) outacc[i] = 0.f;
  if (i < BN_ * HEADS_) { smax[i] = __int_as_float(0xff800000); denom[i] = 0.f; }
}

// Pass A: per (edge, head): score = scale*q[tgt,h].k[src,h] + a1[src,h] + a2[tgt,h] + cbias[h]
__global__ void edge_scores(const int* __restrict__ ei, const float* __restrict__ qkv,
                            const float* __restrict__ cbias, float* __restrict__ scores,
                            float* __restrict__ smax) {
  int gid = blockIdx.x * blockDim.x + threadIdx.x;
  if (gid >= BE_ * HEADS_) return;
  int e = gid >> 3, h = gid & 7;
  int b = e / E_;
  int src = ei[2 * e] + b * N_;
  int tgt = ei[2 * e + 1] + b * N_;
  const float4* q4 = (const float4*)(qkv + (size_t)tgt * NCAT + h * 32);
  const float4* k4 = (const float4*)(qkv + (size_t)src * NCAT + 256 + h * 32);
  float dot = 0.f;
#pragma unroll
  for (int i = 0; i < 8; ++i) {
    float4 qv = q4[i], kv = k4[i];
    dot += qv.x * kv.x + qv.y * kv.y + qv.z * kv.z + qv.w * kv.w;
  }
  float sc = dot * 0.17677669529663687f                       // 32^-0.5
           + qkv[(size_t)src * NCAT + 768 + h]                // a1[src,h]
           + qkv[(size_t)tgt * NCAT + 776 + h]                // a2[tgt,h]
           + cbias[h];
  scores[gid] = sc;
  float* addr = smax + (size_t)tgt * HEADS_ + h;              // ordered-int float max
  if (sc >= 0.f) atomicMax((int*)addr, __float_as_int(sc));
  else           atomicMin((unsigned int*)addr, __float_as_uint(sc));
}

// Pass B: w = exp(score - smax[tgt]); denom[tgt] += w
__global__ void edge_expsum(const int* __restrict__ ei, const float* __restrict__ smax,
                            float* __restrict__ scores, float* __restrict__ denom) {
  int gid = blockIdx.x * blockDim.x + threadIdx.x;
  if (gid >= BE_ * HEADS_) return;
  int e = gid >> 3, h = gid & 7;
  int b = e / E_;
  int tgt = ei[2 * e + 1] + b * N_;
  float w = __expf(scores[gid] - smax[(size_t)tgt * HEADS_ + h]);
  scores[gid] = w;
  atomicAdd(denom + (size_t)tgt * HEADS_ + h, w);
}

// Pass C: outacc[tgt] += (w/denom[tgt]) * v[src]; 64 threads/edge, float4 each
__global__ void edge_scatter(const int* __restrict__ ei, const float* __restrict__ qkv,
                             const float* __restrict__ scores, const float* __restrict__ denom,
                             float* __restrict__ outacc) {
  int gid = blockIdx.x * blockDim.x + threadIdx.x;
  if (gid >= BE_ * 64) return;
  int e = gid >> 6;
  int t = gid & 63;
  int h = t >> 3, dg = t & 7;
  int b = e / E_;
  int src = ei[2 * e] + b * N_;
  int tgt = ei[2 * e + 1] + b * N_;
  float w = scores[(size_t)e * HEADS_ + h] / denom[(size_t)tgt * HEADS_ + h];
  float4 vv = *(const float4*)(qkv + (size_t)src * NCAT + 512 + h * 32 + dg * 4);
  float* o = outacc + (size_t)tgt * 256 + h * 32 + dg * 4;
  atomicAdd(o + 0, w * vv.x);
  atomicAdd(o + 1, w * vv.y);
  atomicAdd(o + 2, w * vv.z);
  atomicAdd(o + 3, w * vv.w);
}

// ---------------------------------------------------------------------------
extern "C" void kernel_launch(void* const* d_in, const int* in_sizes, int n_in,
                              void* d_out, int out_size, void* d_ws, size_t ws_size,
                              hipStream_t stream) {
  const float* x  = (const float*)d_in[0];
  const int*   ei = (const int*)d_in[1];
  const float* Wq = (const float*)d_in[2];
  const float* bq = (const float*)d_in[3];
  const float* Wk = (const float*)d_in[4];
  const float* bk = (const float*)d_in[5];
  const float* Wv = (const float*)d_in[6];
  const float* bv = (const float*)d_in[7];
  const float* We = (const float*)d_in[8];
  const float* be = (const float*)d_in[9];
  const float* Wa = (const float*)d_in[10];
  const float* ba = (const float*)d_in[11];
  const float* Wo = (const float*)d_in[12];
  const float* bo = (const float*)d_in[13];
  float* out = (float*)d_out;
  (void)in_sizes; (void)n_in; (void)out_size; (void)ws_size;

  // Workspace carve (~193 MB total)
  char* ws = (char*)d_ws;
  size_t off = 0;
  auto carve = [&](size_t bytes) -> char* {
    char* p = ws + off;
    off = (off + bytes + 255) & ~(size_t)255;
    return p;
  };
  _Float16* packW  = (_Float16*)carve((size_t)NCAT * 256 * sizeof(_Float16)); // 410 KB
  _Float16* packWo = (_Float16*)carve((size_t)256 * 256 * sizeof(_Float16));  // 131 KB
  float* bcat   = (float*)carve(NCAT * sizeof(float));
  float* cbias  = (float*)carve(8 * sizeof(float));
  float* qkv    = (float*)carve((size_t)BN_ * NCAT * sizeof(float));          // 128 MB
  float* smax   = (float*)carve((size_t)BN_ * HEADS_ * sizeof(float));        // 1.3 MB
  float* denom  = (float*)carve((size_t)BN_ * HEADS_ * sizeof(float));        // 1.3 MB
  float* scores = (float*)carve((size_t)BE_ * HEADS_ * sizeof(float));        // 20.5 MB
  float* outacc = (float*)carve((size_t)BN_ * 256 * sizeof(float));           // 41 MB

  // 1. Repack weights (compose We@Wa, swizzle into WMMA B-fragment layout)
  int tot = NCAT * 256 + 256 * 256;
  repack_weights<<<(tot + 255) / 256, 256, 0, stream>>>(
      Wq, Wk, Wv, We, Wa, Wo, bq, bk, bv, be, ba, packW, packWo, bcat, cbias);

  // 2. Fused [q|k|v|a1|a2|pad] = x @ Wcat + bcat   ([40000,256] x [256,800])
  //    50 N-tiles = 25 pairs; 8 pairs/block -> grid.y = 4
  dim3 g1(BN_ / 16, (NT_CAT / 2 + 7) / 8);
  gemm_xw_wmma<<<g1, 256, 0, stream>>>(x, packW, bcat, qkv, NCAT, NT_CAT / 2);

  // 3. Init segment-softmax state + output accumulator
  init_buffers<<<(BN_ * 256 + 255) / 256, 256, 0, stream>>>(smax, denom, outacc);

  // 4. Edge passes (gathers hit L2-resident node tables)
  edge_scores <<<(BE_ * HEADS_ + 255) / 256, 256, 0, stream>>>(ei, qkv, cbias, scores, smax);
  edge_expsum <<<(BE_ * HEADS_ + 255) / 256, 256, 0, stream>>>(ei, smax, scores, denom);
  edge_scatter<<<(BE_ * 64 + 255) / 256, 256, 0, stream>>>(ei, qkv, scores, denom, outacc);

  // 5. out = outacc @ Wo + bo   ([40000,256] x [256,256]) -> 8 pairs, grid.y = 1
  dim3 g2(BN_ / 16, 1);
  gemm_xw_wmma<<<g2, 256, 0, stream>>>(outacc, packWo, bo, out, 256, 8);
}
